// WindowAttention_16836271801065
// MI455X (gfx1250) — compile-verified, matrix operand order checked by reading, not run on hardware
//
#include <hip/hip_runtime.h>

typedef __attribute__((ext_vector_type(16))) _Float16 v16h;
typedef __attribute__((ext_vector_type(8)))  float    v8f;

#define N_TOK   49
#define NPAD    64
#define HEADS   6
#define C_DIM   192
#define NW      4096
#define NWIN    8192

// LDS layout (bytes)
#define OFF_Q    0                      // f16 [64][192]  Q, later reused as O1 (attn@V)
#define OFF_K    24576                  // f16 [64][192]  K
#define OFF_V    49152                  // f16 [192][64]  V transposed (channel-major)
#define OFF_NRM  73728                  // f32 rq[64][8], rk[64][8]
#define OFF_R    77824                  // f16 X [64][192] (24.5KB), later P [6][64][64] (48KB)
#define SMEM_SZ  (77824 + 49152)        // 126976 B  (2 WGs fit in 320KB WGP LDS)

// Workspace layout (bytes): f16 weights [4][192][192], then bias f32 [6][64][64]
#define WS_BIAS_OFF (4 * C_DIM * C_DIM * 2)

__device__ __forceinline__ v8f vzero() {
  v8f z;
#pragma unroll
  for (int i = 0; i < 8; ++i) z[i] = 0.0f;
  return z;
}

__device__ __forceinline__ v8f wmma16(v16h a, v16h b, v8f c) {
  // D = A(16x32 f16) x B(32x16 f16) + C(16x16 f32)
  return __builtin_amdgcn_wmma_f32_16x16x32_f16(false, a, false, b, (short)0, c,
                                                false, false);
}

// Load a 16x32 f16 fragment from a row-major f16 array (dword pointer).
// Valid as A operand (rows = M, K contiguous) and as B operand when the "row"
// index is the N column and K runs contiguous (i.e. B^T row-major = W row-major
// for X @ W^T). Per ISA layout: lanes 0-15 hold K 0..7 (u0..3) / 16..23 (u4..7);
// lanes 16-31 hold K 8..15 / 24..31. Requires 16B alignment.
__device__ __forceinline__ v16h ld_frag(const unsigned int* p, int m0,
                                        int stride_dw, int k0dw, int lane) {
  const int r = lane & 15, hf = (lane >> 4) & 1;
  const unsigned int* q = p + (m0 + r) * stride_dw + k0dw + hf * 4;
  union { v16h h; uint4 u4[2]; } f;
  f.u4[0] = *(const uint4*)(q);
  f.u4[1] = *(const uint4*)(q + 8);
  return f.h;
}

__device__ __forceinline__ unsigned int pack2h(float a, float b) {
  _Float16 h0 = (_Float16)a, h1 = (_Float16)b;
  return (unsigned int)__builtin_bit_cast(unsigned short, h0) |
         ((unsigned int)__builtin_bit_cast(unsigned short, h1) << 16);
}

// ---------------- pre-kernels ----------------

__global__ void prep_weights(const float* __restrict__ Wq, const float* __restrict__ Wk,
                             const float* __restrict__ Wv, const float* __restrict__ Wp,
                             _Float16* __restrict__ wout) {
  int i = blockIdx.x * blockDim.x + threadIdx.x;       // 4*192*192 = 147456
  int which = i / (C_DIM * C_DIM);
  int j = i % (C_DIM * C_DIM);
  const float* src = (which == 0) ? Wq : (which == 1) ? Wk : (which == 2) ? Wv : Wp;
  wout[i] = (_Float16)src[j];
}

__global__ void prep_bias(const float* __restrict__ rpb, const int* __restrict__ rel,
                          float* __restrict__ biasp) {
  int i = blockIdx.x * blockDim.x + threadIdx.x;       // 6*64*64 = 24576
  int h = i / (NPAD * NPAD);
  int r = (i / NPAD) % NPAD;
  int c = i % NPAD;
  float v = 0.0f;
  if (r < N_TOK && c < N_TOK) v = rpb[rel[r * N_TOK + c] * HEADS + h];
  biasp[i] = v;
}

// ---------------- main kernel: one block per window ----------------

extern "C" __global__ void __launch_bounds__(256)
win_attn_kernel(const float* __restrict__ x, const float* __restrict__ mask,
                const float* __restrict__ d_l, const _Float16* __restrict__ wf16,
                const float* __restrict__ biasp, const float* __restrict__ temp,
                float* __restrict__ out) {
  __shared__ __align__(16) unsigned char smem[SMEM_SZ];

  const int b    = blockIdx.x;
  const int tid  = threadIdx.x;
  const int wave = tid >> 5;
  const int lane = tid & 31;
  const int r    = lane & 15;
  const int hf   = (lane >> 4) & 1;

  _Float16* Qh = (_Float16*)(smem + OFF_Q);     // [64][192], later O1
  _Float16* Kh = (_Float16*)(smem + OFF_K);     // [64][192]
  _Float16* Vt = (_Float16*)(smem + OFF_V);     // [192][64] transposed
  float*    rq = (float*)(smem + OFF_NRM);      // [64][8]
  float*    rk = rq + NPAD * 8;                 // [64][8]
  _Float16* Xh = (_Float16*)(smem + OFF_R);     // [64][192]
  _Float16* Pb = (_Float16*)(smem + OFF_R);     // [6][64][64], aliases Xh

  // ---- stage 1: X -> f16 LDS, packed dword stores; pad rows zero-filled ----
  const float* xb = x + (size_t)b * N_TOK * C_DIM;
  unsigned int* Xst = (unsigned int*)Xh;
  const int NDW = N_TOK * (C_DIM / 2);                 // 4704 data dwords
  for (int i = tid; i < NDW; i += 256) {
    float a0 = xb[2 * i];
    float a1 = xb[2 * i + 1];
    Xst[i] = pack2h(a0, a1);
  }
  for (int i = NDW + tid; i < NPAD * (C_DIM / 2); i += 256) Xst[i] = 0u;
  __syncthreads();

  // ---- stage 2: Q/K/V = X @ W^T. 36 jobs (proj,nt); B-fragments reused
  //      across the 4 M-tiles; proj branch scalarized (wave-uniform). ----
  const unsigned int* Xdw = (const unsigned int*)Xh;
  const unsigned int* Wdw = (const unsigned int*)wf16;   // [4][192][96 dw]
  for (int j = wave; j < 36; j += 8) {
    int p = j / 12, nt = j % 12;
    int ps = __builtin_amdgcn_readfirstlane(p);          // wave-uniform -> s_cbranch
    const unsigned int* Wb = Wdw + p * (C_DIM * 96);
    v16h bfr[6];
#pragma unroll
    for (int kt = 0; kt < 6; ++kt) bfr[kt] = ld_frag(Wb, nt * 16, 96, kt * 16, lane);
    const int col = nt * 16 + r;
    for (int mt = 0; mt < 4; ++mt) {
      v8f acc = vzero();
#pragma unroll
      for (int kt = 0; kt < 6; ++kt)
        acc = wmma16(ld_frag(Xdw, mt * 16, 96, kt * 16, lane), bfr[kt], acc);
      if (ps == 2) {
        // transposed V: the 8 C-elements are consecutive rows -> one b128 store
        union { _Float16 h[8]; uint4 u; } vv;
#pragma unroll
        for (int g = 0; g < 8; ++g) vv.h[g] = (_Float16)acc[g];
        *(uint4*)&Vt[col * NPAD + mt * 16 + 8 * hf] = vv.u;
      } else {
        _Float16* dst = (ps == 0) ? Qh : Kh;
#pragma unroll
        for (int g = 0; g < 8; ++g)
          dst[(mt * 16 + g + 8 * hf) * C_DIM + col] = (_Float16)acc[g];
      }
    }
  }
  __syncthreads();

  // ---- stage 3: per-row/per-head inverse L2 norms of Q and K ----
  for (int i = tid; i < 2 * NPAD * HEADS; i += 256) {
    int isk = (i >= NPAD * HEADS);
    int jj  = isk ? (i - NPAD * HEADS) : i;
    int row = jj / HEADS, h = jj % HEADS;
    const _Float16* src = (isk ? Kh : Qh) + row * C_DIM + h * 32;
    float s = 0.0f;
#pragma unroll
    for (int e = 0; e < 32; ++e) { float v = (float)src[e]; s += v * v; }
    float sc = 1.0f / fmaxf(sqrtf(s), 1e-12f);           // F.normalize eps
    (isk ? rk : rq)[row * 8 + h] = sc;
  }
  __syncthreads();

  // ---- stage 4: logits + bias + mask + d_l, in-register softmax, P -> LDS ----
  const float  dl    = d_l[b];
  const float* maskb = mask + (size_t)(b & (NW - 1)) * N_TOK * N_TOK;
  const unsigned int* Qdw = (const unsigned int*)Qh;
  const unsigned int* Kdw = (const unsigned int*)Kh;
  for (int j = wave; j < HEADS * 4; j += 8) {            // 24 jobs: (head, row-tile)
    int h = j / 4, nt = j % 4;
    float tm = temp[h];
    v16h a = ld_frag(Qdw, nt * 16, 96, h * 16, lane);    // K dim = head dim (32)
    v8f acc[4];
#pragma unroll
    for (int mt = 0; mt < 4; ++mt) {
      v16h bf = ld_frag(Kdw, mt * 16, 96, h * 16, lane);
      acc[mt] = wmma16(a, bf, vzero());
    }
    // hoisted per-row norm scales (fixed per g across mt)
    float rqv[8];
#pragma unroll
    for (int g = 0; g < 8; ++g) rqv[g] = rq[(nt * 16 + g + 8 * hf) * 8 + h] * tm;
    const float* bb = biasp + h * (NPAD * NPAD);
    // fold 1/|q| * 1/|k| * temperature, add bias + mask, scale by d_l
#pragma unroll
    for (int mt = 0; mt < 4; ++mt) {
      int col = mt * 16 + r;
      float rkc = rk[col * 8 + h];
#pragma unroll
      for (int g = 0; g < 8; ++g) {
        int row = nt * 16 + g + 8 * hf;
        float v = acc[mt][g] * rqv[g] * rkc;
        v += bb[row * NPAD + col];
        if (row < N_TOK && col < N_TOK) v += maskb[row * N_TOK + col];
        v *= dl;
        if (col >= N_TOK) v = -1e30f;                    // mask pad columns
        acc[mt][g] = v;
      }
    }
    // softmax per row: a C-row lives in 16 lanes of one half, fixed vgpr g
#pragma unroll
    for (int g = 0; g < 8; ++g) {
      float m = fmaxf(fmaxf(acc[0][g], acc[1][g]), fmaxf(acc[2][g], acc[3][g]));
#pragma unroll
      for (int s = 1; s < 16; s <<= 1) m = fmaxf(m, __shfl_xor(m, s, 32));
      float sum = 0.0f;
#pragma unroll
      for (int mt = 0; mt < 4; ++mt) {
        float e = __expf(acc[mt][g] - m);
        acc[mt][g] = e; sum += e;
      }
#pragma unroll
      for (int s = 1; s < 16; s <<= 1) sum += __shfl_xor(sum, s, 32);
      float inv = 1.0f / sum;
#pragma unroll
      for (int mt = 0; mt < 4; ++mt) acc[mt][g] *= inv;
    }
    // store probabilities f16 (pad cols are exp(-1e30)->0)
#pragma unroll
    for (int mt = 0; mt < 4; ++mt) {
      int col = mt * 16 + r;
#pragma unroll
      for (int g = 0; g < 8; ++g) {
        int row = nt * 16 + g + 8 * hf;
        Pb[h * (NPAD * NPAD) + row * NPAD + col] = (_Float16)acc[mt][g];
      }
    }
  }
  __syncthreads();

  // ---- stage 5: O1 = P @ V (per head), O1 f16 overwrites Q region ----
  _Float16* O1 = Qh;
  const unsigned int* Pdw  = (const unsigned int*)Pb;
  const unsigned int* Vtdw = (const unsigned int*)Vt;
  for (int j = wave; j < HEADS * 4 * 2; j += 8) {        // 48 jobs
    int h = j / 8, rest = j % 8, nt = rest / 2, dt = rest % 2;
    v8f acc = vzero();
#pragma unroll
    for (int kk = 0; kk < 2; ++kk) {                     // K = tokens, 2 x 32
      v16h a  = ld_frag(Pdw + h * 2048, nt * 16, 32, kk * 16, lane);
      v16h bf = ld_frag(Vtdw, h * 32 + dt * 16, 32, kk * 16, lane);
      acc = wmma16(a, bf, acc);
    }
    int col = h * 32 + dt * 16 + r;
#pragma unroll
    for (int g = 0; g < 8; ++g) {
      int row = nt * 16 + g + 8 * hf;
      O1[row * C_DIM + col] = (_Float16)acc[g];
    }
  }
  __syncthreads();

  // ---- stage 6: out = O1 @ Wp^T; 24 jobs (nt, m-half), B reused over 2 M-tiles ----
  const unsigned int* O1dw = (const unsigned int*)O1;
  const unsigned int* Wpdw = Wdw + 3 * (C_DIM * 96);
  float* outb = out + (size_t)b * N_TOK * C_DIM;
  for (int j = wave; j < 24; j += 8) {
    int nt = j / 2, mh = j % 2;
    v16h bfr[6];
#pragma unroll
    for (int kt = 0; kt < 6; ++kt) bfr[kt] = ld_frag(Wpdw, nt * 16, 96, kt * 16, lane);
    const int col = nt * 16 + r;
    for (int mq = 0; mq < 2; ++mq) {
      int mt = mh * 2 + mq;
      v8f acc = vzero();
#pragma unroll
      for (int kt = 0; kt < 6; ++kt)
        acc = wmma16(ld_frag(O1dw, mt * 16, 96, kt * 16, lane), bfr[kt], acc);
#pragma unroll
      for (int g = 0; g < 8; ++g) {
        int row = mt * 16 + g + 8 * hf;
        if (row < N_TOK) outb[row * C_DIM + col] = acc[g];
      }
    }
  }
}

// ---------------- launch ----------------

extern "C" void kernel_launch(void* const* d_in, const int* in_sizes, int n_in,
                              void* d_out, int out_size, void* d_ws, size_t ws_size,
                              hipStream_t stream) {
  (void)in_sizes; (void)n_in; (void)out_size; (void)ws_size;
  const float* x    = (const float*)d_in[0];
  const float* mask = (const float*)d_in[1];
  const float* d_l  = (const float*)d_in[2];
  const float* Wq   = (const float*)d_in[3];
  const float* Wk   = (const float*)d_in[4];
  const float* Wv   = (const float*)d_in[5];
  const float* Wp   = (const float*)d_in[6];
  const float* temp = (const float*)d_in[7];
  const float* rpb  = (const float*)d_in[8];
  const int*   rel  = (const int*)d_in[9];
  float* out = (float*)d_out;

  _Float16* wf16   = (_Float16*)d_ws;
  float*    biasws = (float*)((char*)d_ws + WS_BIAS_OFF);

  prep_weights<<<(4 * C_DIM * C_DIM) / 256, 256, 0, stream>>>(Wq, Wk, Wv, Wp, wf16);
  prep_bias<<<(HEADS * NPAD * NPAD) / 256, 256, 0, stream>>>(rpb, rel, biasws);
  win_attn_kernel<<<NWIN, 256, 0, stream>>>(x, mask, d_l, wf16, biasws, temp, out);
}
